// FastASLGNN_71700184040114
// MI455X (gfx1250) — matrix operand order repaired and checked by reading, not compiled, not run on hardware
//
#include <hip/hip_runtime.h>

// CDNA5 / gfx1250: wave32, WMMA 16x16x32 f16->f32.
typedef __attribute__((ext_vector_type(16))) _Float16 v16h;
typedef __attribute__((ext_vector_type(8)))  float    v8f;

#define TB 256

// ---------------- generic fill ----------------
__global__ void k_fill(float* __restrict__ p, float v, size_t n) {
  size_t i = (size_t)blockIdx.x * blockDim.x + threadIdx.x;
  size_t stride = (size_t)gridDim.x * blockDim.x;
  for (; i < n; i += stride) p[i] = v;
}

// ---------------- degree: deg[i] = 1 + #in-edges ----------------
__global__ void k_deg(const int* __restrict__ dst, float* __restrict__ deg, int E) {
  int e = blockIdx.x * blockDim.x + threadIdx.x;
  if (e < E) atomicAdd(&deg[dst[e]], 1.0f);
}

__global__ void k_rsqrt_inplace(float* __restrict__ p, int n) {
  int i = blockIdx.x * blockDim.x + threadIdx.x;
  if (i < n) p[i] = rsqrtf(p[i]);   // deg -> 1/sqrt(deg)
}

// ---------------- WMMA GEMM (compile-time shapes -> fully unrolled, branchless) ------------
// out[M,NCOLS] = op(A[M,K] @ W[K,NCOLS]).  One wave: 16 rows x NT*16 cols, K chunks of 32.
// M must be a multiple of 16 (true for all call sites: 400000, 8192).
template <int K, int NCOLS, int LDA, int LDOUT, bool RELU, bool HASBIAS>
__global__ __launch_bounds__(128) void k_gemm_wmma(const float* __restrict__ A,
                                                   const float* __restrict__ W,
                                                   const float* __restrict__ bias,
                                                   float* __restrict__ out, int M) {
  constexpr int NT = (NCOLS + 15) / 16;
  constexpr int KC = (K + 31) / 32;
  constexpr bool KPAD = (K % 32) != 0;
  constexpr bool NPAD = (NCOLS % 16) != 0;

  int wave = (blockIdx.x * blockDim.x + threadIdx.x) >> 5;
  if (wave >= (M >> 4)) return;             // wave-uniform branch: EXEC all-1s at WMMA
  int lane = threadIdx.x & 31;
  int m0   = wave << 4;
  int mrow = lane & 15;
  int hi   = lane >> 4;                     // half-wave select
  int nlo  = lane & 15;
  const float* Arow = A + (size_t)(m0 + mrow) * LDA;

  v8f acc[NT];
  #pragma unroll
  for (int i = 0; i < NT; ++i) acc[i] = v8f{};

  #pragma unroll
  for (int kc = 0; kc < KC; ++kc) {
    // A fragment: 16-bit A 16x32 layout (ISA 7.12.2):
    // lane L, vgpr v holds k = (v/4)*16 + (L/16)*8 + (v%4)*2, k+1
    v16h a;
    #pragma unroll
    for (int v = 0; v < 8; ++v) {
      int k = kc * 32 + ((v >> 2) << 4) + (hi << 3) + ((v & 3) << 1);
      float f0, f1;
      if constexpr (!KPAD) {
        f0 = Arow[k];
        f1 = Arow[k + 1];
      } else {
        int kk = (k <= K - 2) ? k : (K - 2);   // clamp: always-valid address
        f0 = Arow[kk];
        f1 = Arow[kk + 1];
        if (k     >= K) f0 = 0.0f;             // branchless v_cndmask
        if (k + 1 >= K) f1 = 0.0f;
      }
      a[2 * v]     = (_Float16)f0;
      a[2 * v + 1] = (_Float16)f1;
    }
    #pragma unroll
    for (int nt = 0; nt < NT; ++nt) {
      int n = (nt << 4) + nlo;
      // B fragment: 32x16, col n = lane%16; vgpr v holds k = (v/4)*16 + (L/16)*8 + (v%4)*2, k+1
      v16h b;
      #pragma unroll
      for (int v = 0; v < 8; ++v) {
        int k = kc * 32 + ((v >> 2) << 4) + (hi << 3) + ((v & 3) << 1);
        float f0, f1;
        if constexpr (!KPAD && !NPAD) {
          f0 = W[(size_t)k * NCOLS + n];
          f1 = W[(size_t)(k + 1) * NCOLS + n];
        } else {
          int kk = (k <= K - 2) ? k : (K - 2);
          int nn = (n < NCOLS) ? n : (NCOLS - 1);
          f0 = W[(size_t)kk * NCOLS + nn];
          f1 = W[(size_t)(kk + 1) * NCOLS + nn];
          bool z0 = (KPAD && k >= K) || (NPAD && n >= NCOLS);
          bool z1 = (KPAD && k + 1 >= K) || (NPAD && n >= NCOLS);
          if (z0) f0 = 0.0f;
          if (z1) f1 = 0.0f;
        }
        b[2 * v]     = (_Float16)f0;
        b[2 * v + 1] = (_Float16)f1;
      }
      acc[nt] = __builtin_amdgcn_wmma_f32_16x16x32_f16(
          /*neg_a=*/false, a, /*neg_b=*/false, b,
          /*c_mod=*/(short)0, acc[nt], /*reuse_a=*/false, /*reuse_b=*/false);
    }
  }

  // C/D layout: vgpr v -> M = v + 8*hi, N = lane%16
  #pragma unroll
  for (int nt = 0; nt < NT; ++nt) {
    int n = (nt << 4) + nlo;
    bool nok = !NPAD || (n < NCOLS);
    float bv = 0.0f;
    if constexpr (HASBIAS) {
      int nn = NPAD ? ((n < NCOLS) ? n : (NCOLS - 1)) : n;
      bv = bias[nn];
    }
    float* orow = out + (size_t)(m0 + (hi << 3)) * LDOUT + n;
    #pragma unroll
    for (int v = 0; v < 8; ++v) {
      float val = acc[nt][v] + bv;
      if constexpr (RELU) val = fmaxf(val, 0.0f);
      if (nok) orow[(size_t)v * LDOUT] = val;
    }
  }
}

// ---------------- agg init: agg = hw * (1/deg) + b  (self-loop term + bias) ----------------
__global__ void k_agg_init(const float* __restrict__ hw, const float* __restrict__ dinv,
                           const float* __restrict__ b, float* __restrict__ agg, int N) {
  int t = blockIdx.x * blockDim.x + threadIdx.x;
  if (t >= N * 16) return;
  int node = t >> 4;
  int c0 = (t & 15) << 2;
  float di = dinv[node];
  float invdeg = di * di;                 // dinv = rsqrt(deg) -> dinv^2 = 1/deg
  const float4 v = *(const float4*)(hw + (size_t)node * 64 + c0);
  float4 o;
  o.x = v.x * invdeg + b[c0];
  o.y = v.y * invdeg + b[c0 + 1];
  o.z = v.z * invdeg + b[c0 + 2];
  o.w = v.w * invdeg + b[c0 + 3];
  *(float4*)(agg + (size_t)node * 64 + c0) = o;
}

// ---------------- edge scatter: agg[dst] += hw[src] * dinv[src]*dinv[dst] ----------------
__global__ void k_scatter(const float* __restrict__ hw, const int* __restrict__ src,
                          const int* __restrict__ dst, const float* __restrict__ dinv,
                          float* __restrict__ agg, int E) {
  int t = blockIdx.x * blockDim.x + threadIdx.x;
  if (t >= E * 16) return;
  int e = t >> 4;
  int c0 = (t & 15) << 2;
  int s = src[e], d = dst[e];
  float coef = dinv[s] * dinv[d];
  const float4 v = *(const float4*)(hw + (size_t)s * 64 + c0);
  float* ap = agg + (size_t)d * 64 + c0;
  atomicAdd(ap + 0, v.x * coef);
  atomicAdd(ap + 1, v.y * coef);
  atomicAdd(ap + 2, v.z * coef);
  atomicAdd(ap + 3, v.w * coef);
}

// ---------------- batchnorm stats: stats[c]=sum, stats[64+c]=sumsq ----------------
__global__ void k_bn_stats(const float* __restrict__ t, float* __restrict__ stats, int N) {
  int tid = blockIdx.x * blockDim.x + threadIdx.x;
  int c = tid & 63;
  int r = tid >> 6;
  int stride = (gridDim.x * blockDim.x) >> 6;
  float s = 0.0f, s2 = 0.0f;
  for (; r < N; r += stride) {
    float v = t[(size_t)r * 64 + c];
    s += v; s2 += v * v;
  }
  atomicAdd(&stats[c], s);
  atomicAdd(&stats[64 + c], s2);
}

// ---------------- normalize + affine + relu ----------------
__global__ void k_bn_norm(const float* __restrict__ t, const float* __restrict__ stats,
                          const float* __restrict__ g, const float* __restrict__ beta,
                          float* __restrict__ h, int N) {
  int tid = blockIdx.x * blockDim.x + threadIdx.x;
  if (tid >= N * 16) return;
  int node = tid >> 4;
  int c0 = (tid & 15) << 2;
  float invN = 1.0f / (float)N;
  float4 v = *(const float4*)(t + (size_t)node * 64 + c0);
  float4 o;
  {
    float mu = stats[c0] * invN;
    float var = stats[64 + c0] * invN - mu * mu;
    o.x = fmaxf((v.x - mu) * rsqrtf(var + 1e-5f) * g[c0] + beta[c0], 0.0f);
  }
  {
    int c = c0 + 1; float mu = stats[c] * invN;
    float var = stats[64 + c] * invN - mu * mu;
    o.y = fmaxf((v.y - mu) * rsqrtf(var + 1e-5f) * g[c] + beta[c], 0.0f);
  }
  {
    int c = c0 + 2; float mu = stats[c] * invN;
    float var = stats[64 + c] * invN - mu * mu;
    o.z = fmaxf((v.z - mu) * rsqrtf(var + 1e-5f) * g[c] + beta[c], 0.0f);
  }
  {
    int c = c0 + 3; float mu = stats[c] * invN;
    float var = stats[64 + c] * invN - mu * mu;
    o.w = fmaxf((v.w - mu) * rsqrtf(var + 1e-5f) * g[c] + beta[c], 0.0f);
  }
  *(float4*)(h + (size_t)node * 64 + c0) = o;
}

// ---------------- pooling ----------------
__global__ void k_cnt(const int* __restrict__ batch, float* __restrict__ cnt, int N) {
  int i = blockIdx.x * blockDim.x + threadIdx.x;
  if (i < N) atomicAdd(&cnt[batch[i]], 1.0f);
}

// h >= 0 after relu, so float max == unsigned-bit max with 0 init.
__global__ void k_pool(const float* __restrict__ h, const int* __restrict__ batch,
                       float* __restrict__ gsum, unsigned* __restrict__ gmax, int N) {
  int t = blockIdx.x * blockDim.x + threadIdx.x;
  if (t >= N * 16) return;
  int node = t >> 4;
  int c0 = (t & 15) << 2;
  int g = batch[node];
  float4 v = *(const float4*)(h + (size_t)node * 64 + c0);
  float* sp = gsum + (size_t)g * 64 + c0;
  unsigned* mp = gmax + (size_t)g * 64 + c0;
  atomicAdd(sp + 0, v.x); atomicMax(mp + 0, __float_as_uint(v.x));
  atomicAdd(sp + 1, v.y); atomicMax(mp + 1, __float_as_uint(v.y));
  atomicAdd(sp + 2, v.z); atomicMax(mp + 2, __float_as_uint(v.z));
  atomicAdd(sp + 3, v.w); atomicMax(mp + 3, __float_as_uint(v.w));
}

__global__ void k_xg(const float* __restrict__ gsum, const unsigned* __restrict__ gmax,
                     const float* __restrict__ cnt, float* __restrict__ xg, int G) {
  int t = blockIdx.x * blockDim.x + threadIdx.x;
  if (t >= G * 16) return;
  int g = t >> 4;
  int c0 = (t & 15) << 2;
  float inv = 1.0f / cnt[g];
  float4 s = *(const float4*)(gsum + (size_t)g * 64 + c0);
  float4 mean = {s.x * inv, s.y * inv, s.z * inv, s.w * inv};
  float4 mx;
  mx.x = __uint_as_float(gmax[(size_t)g * 64 + c0 + 0]);
  mx.y = __uint_as_float(gmax[(size_t)g * 64 + c0 + 1]);
  mx.z = __uint_as_float(gmax[(size_t)g * 64 + c0 + 2]);
  mx.w = __uint_as_float(gmax[(size_t)g * 64 + c0 + 3]);
  *(float4*)(xg + (size_t)g * 128 + c0) = mean;
  *(float4*)(xg + (size_t)g * 128 + 64 + c0) = mx;
}

extern "C" void kernel_launch(void* const* d_in, const int* in_sizes, int n_in,
                              void* d_out, int out_size, void* d_ws, size_t ws_size,
                              hipStream_t stream) {
  const float* x     = (const float*)d_in[0];
  const int*   eidx  = (const int*)d_in[1];
  const int*   batch = (const int*)d_in[2];
  const float* W_in  = (const float*)d_in[4];
  const float* b_in  = (const float*)d_in[5];
  const float* Wl[3]    = {(const float*)d_in[6],  (const float*)d_in[10], (const float*)d_in[14]};
  const float* bl[3]    = {(const float*)d_in[7],  (const float*)d_in[11], (const float*)d_in[15]};
  const float* gl[3]    = {(const float*)d_in[8],  (const float*)d_in[12], (const float*)d_in[16]};
  const float* betal[3] = {(const float*)d_in[9],  (const float*)d_in[13], (const float*)d_in[17]};
  const float* Wc1 = (const float*)d_in[18];
  const float* bc1 = (const float*)d_in[19];
  const float* Wc2 = (const float*)d_in[20];
  const float* bc2 = (const float*)d_in[21];
  float* out = (float*)d_out;

  const int N = in_sizes[2];
  const int E = in_sizes[1] / 2;
  const int C = 29;
  const int G = out_size / C;          // 8192
  const int* src = eidx;
  const int* dst = eidx + E;

  // ---- carve workspace (256B aligned slabs) ----
  char* wsp = (char*)d_ws;
  auto carve = [&](size_t bytes) {
    char* p = wsp;
    wsp += (bytes + 255) & ~(size_t)255;
    return p;
  };
  float*    h      = (float*)   carve((size_t)N * 64 * 4);
  float*    hw     = (float*)   carve((size_t)N * 64 * 4);
  float*    agg    = (float*)   carve((size_t)N * 64 * 4);
  float*    dinv   = (float*)   carve((size_t)N * 4);
  float*    stats  = (float*)   carve(128 * 4);
  float*    gsum   = (float*)   carve((size_t)G * 64 * 4);
  unsigned* gmax   = (unsigned*)carve((size_t)G * 64 * 4);
  float*    cnt    = (float*)   carve((size_t)G * 4);
  float*    xg     = (float*)   carve((size_t)G * 128 * 4);
  float*    hidden = (float*)   carve((size_t)G * 64 * 4);
  (void)ws_size; (void)n_in;

  auto gblocks = [](int M) { return ((M >> 4) + 3) / 4; };   // 4 waves / 128-thread block

  // ---- degrees (shared by all 3 GCN layers) ----
  k_fill<<<1024, TB, 0, stream>>>(dinv, 1.0f, (size_t)N);
  k_deg<<<(E + TB - 1) / TB, TB, 0, stream>>>(dst, dinv, E);
  k_rsqrt_inplace<<<(N + TB - 1) / TB, TB, 0, stream>>>(dinv, N);

  // ---- input layer: h = relu(x @ W_in + b_in), K=6 zero-padded ----
  k_gemm_wmma<6, 64, 6, 64, true, true>
      <<<gblocks(N), 128, 0, stream>>>(x, W_in, b_in, h, N);

  // ---- 3 GCN layers ----
  for (int l = 0; l < 3; ++l) {
    k_gemm_wmma<64, 64, 64, 64, false, false>
        <<<gblocks(N), 128, 0, stream>>>(h, Wl[l], nullptr, hw, N);
    k_agg_init<<<(N * 16 + TB - 1) / TB, TB, 0, stream>>>(hw, dinv, bl[l], agg, N);
    k_scatter<<<(E * 16 + TB - 1) / TB, TB, 0, stream>>>(hw, src, dst, dinv, agg, E);
    k_fill<<<1, 128, 0, stream>>>(stats, 0.0f, (size_t)128);
    k_bn_stats<<<512, 256, 0, stream>>>(agg, stats, N);
    k_bn_norm<<<(N * 16 + TB - 1) / TB, TB, 0, stream>>>(agg, stats, gl[l], betal[l], h, N);
  }

  // ---- pooling ----
  k_fill<<<256, TB, 0, stream>>>(gsum, 0.0f, (size_t)G * 64);
  k_fill<<<256, TB, 0, stream>>>((float*)gmax, 0.0f, (size_t)G * 64);  // 0u bits == 0.0f
  k_fill<<<32, TB, 0, stream>>>(cnt, 0.0f, (size_t)G);
  k_cnt<<<(N + TB - 1) / TB, TB, 0, stream>>>(batch, cnt, N);
  k_pool<<<(N * 16 + TB - 1) / TB, TB, 0, stream>>>(h, batch, gsum, gmax, N);
  k_xg<<<(G * 16 + TB - 1) / TB, TB, 0, stream>>>(gsum, gmax, cnt, xg, G);

  // ---- classifier head ----
  k_gemm_wmma<128, 64, 128, 64, true, true>
      <<<gblocks(G), 128, 0, stream>>>(xg, Wc1, bc1, hidden, G);
  k_gemm_wmma<64, 29, 64, 29, false, true>
      <<<gblocks(G), 128, 0, stream>>>(hidden, Wc2, bc2, out, G);
}
